// GCNLayer_82721070121689
// MI455X (gfx1250) — matrix operand order repaired and checked by reading, not compiled, not run on hardware
//
#include <hip/hip_runtime.h>

// GCN layer: gather -> segment-mean (atomics, L2-resident) -> split-fp32 bf16 WMMA GEMM.
// MI455X / gfx1250, wave32.

#define DEV static __device__ __forceinline__

constexpr int N_NODES = 10000;
constexpr int N_EDGES = 640000;
constexpr int IN_F    = 128;
constexpr int OUT_F   = 512;

typedef __attribute__((ext_vector_type(16))) __bf16       v16bf;
typedef __attribute__((ext_vector_type(8)))  float        v8f;
typedef __attribute__((ext_vector_type(4)))  unsigned int v4u;
typedef __attribute__((ext_vector_type(4)))  float        v4f;

union Frag16 { v16bf v; v4u q[2]; };

DEV unsigned short f32_to_bf16_rne(float f) {
  unsigned int u = __float_as_uint(f);
  u += 0x7fffu + ((u >> 16) & 1u);      // round to nearest even
  return (unsigned short)(u >> 16);
}
DEV float bf16_to_f32(unsigned short h) {
  return __uint_as_float(((unsigned int)h) << 16);
}

// ---------------- Phase 0: zero accumulators (ws is poisoned 0xAA) ----------
__global__ void k_init(float* __restrict__ sums, float* __restrict__ deg) {
  int i = blockIdx.x * 256 + threadIdx.x;
  if (i < N_NODES * IN_F) sums[i] = 0.0f;
  if (i < N_NODES)        deg[i]  = 0.0f;
}

// ---------------- Phase 1: edge scatter (one wave per edge) -----------------
// lane l handles features [4l, 4l+4): float4 gather + 4 hardware f32 atomics.
__global__ void k_scatter(const float* __restrict__ x, const int* __restrict__ src,
                          const int* __restrict__ dst, float* __restrict__ sums,
                          float* __restrict__ deg) {
  int t    = blockIdx.x * 256 + threadIdx.x;     // E*32 threads, exact multiple
  int e    = t >> 5;
  int lane = t & 31;
  int s = src[e] * IN_F;
  int d = dst[e] * IN_F;
  v4f m = *(const v4f*)(x + s + lane * 4);
  float* p = sums + d + lane * 4;
  unsafeAtomicAdd(p + 0, m.x);
  unsafeAtomicAdd(p + 1, m.y);
  unsafeAtomicAdd(p + 2, m.z);
  unsafeAtomicAdd(p + 3, m.w);
  if (lane == 0) unsafeAtomicAdd(deg + (d >> 7), 1.0f);
}

// ---------------- Phase 2: mean / fallback + split h into bf16 hi/lo --------
__global__ void k_finalize(const float* __restrict__ x, const float* __restrict__ sums,
                           const float* __restrict__ deg,
                           unsigned short* __restrict__ hHi, unsigned short* __restrict__ hLo) {
  int i = blockIdx.x * 256 + threadIdx.x;        // N*F threads, exact multiple
  float d = deg[i >> 7];
  float h = (d > 0.0f) ? (sums[i] / d) : x[i];
  unsigned short hi = f32_to_bf16_rne(h);
  hHi[i] = hi;
  hLo[i] = f32_to_bf16_rne(h - bf16_to_f32(hi));
}

// ---------------- Phase 3: split W into bf16 hi/lo, transposed [N][K] -------
__global__ void k_splitW(const float* __restrict__ W,
                         unsigned short* __restrict__ WhiT, unsigned short* __restrict__ WloT) {
  int i = blockIdx.x * 256 + threadIdx.x;        // 128*512 threads
  int k = i >> 9;
  int n = i & 511;
  float w = W[i];
  unsigned short hi = f32_to_bf16_rne(w);
  WhiT[n * IN_F + k] = hi;
  WloT[n * IN_F + k] = f32_to_bf16_rne(w - bf16_to_f32(hi));
}

// ---------------- Phase 4: WMMA GEMM, split-fp32 (hi*hi + hi*lo + lo*hi) ----
// 256 threads = 8 waves; wave w computes the 16x16 tile (blockIdx.x, blockIdx.y*8+w).
// All dims divide exactly -> no predication, EXEC all-ones for every WMMA.
__global__ __launch_bounds__(256)
void k_gemm(const unsigned short* __restrict__ hHi, const unsigned short* __restrict__ hLo,
            const unsigned short* __restrict__ WhiT, const unsigned short* __restrict__ WloT,
            const float* __restrict__ bias, float* __restrict__ out) {
  const int wave  = threadIdx.x >> 5;
  const int lane  = threadIdx.x & 31;
  const int half  = lane >> 4;                   // 0 or 1
  const int l16   = lane & 15;
  const int mTile = blockIdx.x;                  // 0..624
  const int nTile = blockIdx.y * 8 + wave;       // 0..31

  const int row = mTile * 16 + l16;              // A row for this lane
  const int col = nTile * 16 + l16;              // B/D column for this lane

  // A (16x32 bf16): slots 0-7 = K kk+8*half.., slots 8-15 = +16  -> two 16B loads
  const unsigned short* aHiP = hHi  + row * IN_F + 8 * half;
  const unsigned short* aLoP = hLo  + row * IN_F + 8 * half;
  // B (32x16 bf16): slots 0-7 = K kk+8*half.., slots 8-15 = +16  -> two 16B loads
  const unsigned short* bHiP = WhiT + col * IN_F + 8 * half;
  const unsigned short* bLoP = WloT + col * IN_F + 8 * half;

  v8f acc = {};
#pragma unroll
  for (int kk = 0; kk < IN_F; kk += 32) {
    Frag16 aHi, aLo, bHi, bLo;
    aHi.q[0] = *(const v4u*)(aHiP + kk);
    aHi.q[1] = *(const v4u*)(aHiP + kk + 16);
    aLo.q[0] = *(const v4u*)(aLoP + kk);
    aLo.q[1] = *(const v4u*)(aLoP + kk + 16);
    bHi.q[0] = *(const v4u*)(bHiP + kk);
    bHi.q[1] = *(const v4u*)(bHiP + kk + 16);
    bLo.q[0] = *(const v4u*)(bLoP + kk);
    bLo.q[1] = *(const v4u*)(bLoP + kk + 16);
    acc = __builtin_amdgcn_wmma_f32_16x16x32_bf16(false, aHi.v, false, bHi.v,
                                                  (short)0, acc, false, false);
    acc = __builtin_amdgcn_wmma_f32_16x16x32_bf16(false, aHi.v, false, bLo.v,
                                                  (short)0, acc, false, false);
    acc = __builtin_amdgcn_wmma_f32_16x16x32_bf16(false, aLo.v, false, bHi.v,
                                                  (short)0, acc, false, false);
  }

  // D layout: VGPR v -> row mTile*16 + v + 8*half, column = col. Add bias, store.
  const float bv = bias[col];
  float* o = out + (mTile * 16 + 8 * half) * OUT_F + col;
#pragma unroll
  for (int v = 0; v < 8; ++v)
    o[v * OUT_F] = acc[v] + bv;
}

extern "C" void kernel_launch(void* const* d_in, const int* in_sizes, int n_in,
                              void* d_out, int out_size, void* d_ws, size_t ws_size,
                              hipStream_t stream) {
  (void)in_sizes; (void)n_in; (void)out_size; (void)ws_size;
  const float* x   = (const float*)d_in[0];
  const int*   src = (const int*)d_in[1];
  const int*   dst = (const int*)d_in[2];
  const float* W   = (const float*)d_in[3];
  const float* b   = (const float*)d_in[4];
  float*       out = (float*)d_out;

  // Workspace carve-out (~10.5 MB), 256B-aligned slabs.
  char* ws = (char*)d_ws;
  size_t off = 0;
  auto carve = [&](size_t bytes) -> void* {
    void* p = ws + off;
    off += (bytes + 255) & ~size_t(255);
    return p;
  };
  float*          sums = (float*)carve((size_t)N_NODES * IN_F * sizeof(float));
  float*          deg  = (float*)carve((size_t)N_NODES * sizeof(float));
  unsigned short* hHi  = (unsigned short*)carve((size_t)N_NODES * IN_F * 2);
  unsigned short* hLo  = (unsigned short*)carve((size_t)N_NODES * IN_F * 2);
  unsigned short* WhiT = (unsigned short*)carve((size_t)IN_F * OUT_F * 2);
  unsigned short* WloT = (unsigned short*)carve((size_t)IN_F * OUT_F * 2);

  k_init    <<<(N_NODES * IN_F + 255) / 256, 256, 0, stream>>>(sums, deg);
  k_scatter <<<(N_EDGES * 32) / 256,        256, 0, stream>>>(x, src, dst, sums, deg);
  k_finalize<<<(N_NODES * IN_F) / 256,      256, 0, stream>>>(x, sums, deg, hHi, hLo);
  k_splitW  <<<(IN_F * OUT_F) / 256,        256, 0, stream>>>(W, WhiT, WloT);
  k_gemm    <<<dim3(N_NODES / 16, OUT_F / 128), 256, 0, stream>>>(hHi, hLo, WhiT, WloT, b, out);
}
